// DilatedResidualBlock_87943750352952
// MI455X (gfx1250) — matrix operand order repaired and checked by reading, not compiled, not run on hardware
//
#include <hip/hip_runtime.h>
#include <hip/hip_bf16.h>

// ---------------------------------------------------------------------------
// DilatedResidualBlock (RandLA-Net style LocSE + attentive pooling) for
// MI455X / gfx1250.  All matmuls run on v_wmma_f32_16x16x32_f16: the K=16
// neighbors of one point form exactly one 16-row WMMA M-tile.
//   pass 1: brute-force KNN (LDS-tiled + global_prefetch, register top-16)
//   pass 2: LocSE (2 WMMA matmuls + folded-BN/ReLU) -> concat[B,N,16,128] f16
//           plus the shortcut branch (VALU matvec + folded BN)
//   pass 3/4: softmax-over-N stats: recompute logits (WMMA, w_score in LDS)
//           -> keyed atomicMax then atomicAdd-of-exp into [B,K,128] columns
//   pass 5: TDM tensor_load_to_lds stages each concat tile, recompute logits
//           (WMMA), scores, pooled, att matmul (WMMA), BN+shortcut+ReLU
// ---------------------------------------------------------------------------

#define BB_ 2
#define NN_ 8192
#define KK_ 16
#define EPS_ 1e-5f

typedef __attribute__((ext_vector_type(16))) _Float16 v16h;
typedef __attribute__((ext_vector_type(8)))  float    v8f;
typedef unsigned int u32x4 __attribute__((ext_vector_type(4)));
typedef int          i32x4 __attribute__((ext_vector_type(4)));
typedef int          i32x8 __attribute__((ext_vector_type(8)));

#if defined(__has_include)
#  if __has_include(<hip/amd_detail/amd_gfx1250_TDM.h>)
#    define TDM_ARITY6 1
#  endif
#endif

// ---- monotone float <-> unsigned key (for atomic max over floats) ----------
__device__ __forceinline__ unsigned f2k(float f) {
  unsigned u = __float_as_uint(f);
  return (u & 0x80000000u) ? ~u : (u | 0x80000000u);
}
__device__ __forceinline__ float k2f(unsigned k) {
  unsigned u = (k & 0x80000000u) ? (k & 0x7fffffffu) : ~k;
  return __uint_as_float(u);
}

// ---- WMMA fragment loaders (ISA 7.12.2 layouts, wave32) --------------------
// A 16x32 f16: lane L -> row M=L&15; VGPR v packs K = 2*(v&3) + 16*(v>=4) + 8*(L>=16)
__device__ __forceinline__ v16h load_frag_a(const _Float16* base, int stride) {
  int lane = threadIdx.x & 31;
  int M = lane & 15, hl = lane >> 4;
  v16h a;
#pragma unroll
  for (int r = 0; r < 8; ++r) {
    int k0 = 2 * (r & 3) + ((r >> 2) << 4) + (hl << 3);
    union { unsigned u; _Float16 h[2]; } cv;
    cv.u = *(const unsigned*)(base + M * stride + k0);
    a[2 * r] = cv.h[0];
    a[2 * r + 1] = cv.h[1];
  }
  return a;
}
// B 32x16 f16 from [n][k] (k contiguous): lane L -> col n=L&15; VGPR v packs
// K = 2*(v&3) + 16*(v>=4) + 8*(L>=16)
__device__ __forceinline__ v16h load_frag_b(const _Float16* base, int stride) {
  int lane = threadIdx.x & 31;
  int nn = lane & 15, hl = lane >> 4;
  v16h b;
#pragma unroll
  for (int r = 0; r < 8; ++r) {
    int k0 = 2 * (r & 3) + ((r >> 2) << 4) + (hl << 3);
    union { unsigned u; _Float16 h[2]; } cv;
    cv.u = *(const unsigned*)(base + nn * stride + k0);
    b[2 * r] = cv.h[0];
    b[2 * r + 1] = cv.h[1];
  }
  return b;
}
__device__ __forceinline__ v8f wmma_f16(v16h a, v16h b, v8f c) {
  return __builtin_amdgcn_wmma_f32_16x16x32_f16(false, a, false, b, (short)0, c,
                                                false, false);
}

// ---- Tensor Data Mover: 1-D f16 tile DMA global -> LDS ---------------------
// D# per ISA 8.3-8.5: group0 = {count=1, lds_addr, global_addr[56:0], type=2};
// group1 = {data_size=2B, tensor_dim0=nelem, tile_dim0=nelem (1-D tile),
//           tensor_dim0_stride=nelem}; groups 2/3 zero (<=2-D tensor).
__device__ __forceinline__ void tdm_load_tile_f16(void* lds_ptr,
                                                  const void* gptr,
                                                  unsigned nelem) {
#if __has_builtin(__builtin_amdgcn_tensor_load_to_lds)
  unsigned lds_addr = (unsigned)(unsigned long long)lds_ptr;  // low 32b = LDS offset
  unsigned long long ga = (unsigned long long)gptr;
  u32x4 g0;
  g0.x = 1u;                                            // count=1, user mode
  g0.y = lds_addr;
  g0.z = (unsigned)(ga & 0xffffffffu);                  // global_addr[31:0]
  g0.w = (unsigned)((ga >> 32) & 0x01ffffffu) | (2u << 30);  // addr[56:32]|type=2
  i32x8 g1;
  g1[0] = (1 << 16);                                    // data_size = 2 bytes
  g1[1] = (int)((nelem & 0xffffu) << 16);               // tensor_dim0[15:0] @63:48
  g1[2] = (int)((nelem >> 16) & 0xffffu);               // tensor_dim0[31:16] @79:64
  g1[3] = (int)((nelem & 0xffffu) << 16);               // tile_dim0 @127:112
  g1[4] = 0;                                            // tile_dim1/2 = 0
  g1[5] = (int)nelem;                                   // tensor_dim0_stride
  g1[6] = 0;
  g1[7] = 0;
  i32x4 z4 = {0, 0, 0, 0};
#ifdef TDM_ARITY6
  i32x8 z8 = {0, 0, 0, 0, 0, 0, 0, 0};
  __builtin_amdgcn_tensor_load_to_lds(g0, g1, z4, z4, z8, 0);
#else
  __builtin_amdgcn_tensor_load_to_lds(g0, g1, z4, z4, 0);
#endif
  __builtin_amdgcn_s_wait_tensorcnt(0);
#else
  // host pass / no-TDM fallback: plain vector copy through VGPRs
  int lane = threadIdx.x & 31;
  const uint4* src = (const uint4*)gptr;
  uint4* dst = (uint4*)lds_ptr;
  for (int t = 0; t < (int)(nelem / 8) / 32; ++t)
    dst[lane + t * 32] = src[lane + t * 32];
#endif
}

// ---------------------------------------------------------------------------
// init: colmax = key(-inf), colsum = 0; w_score/w_att -> f16; fold all BNs.
// grid 64x256 (16384 threads)
__global__ void k_init(unsigned* __restrict__ colmax, float* __restrict__ colsum,
                       const float* __restrict__ w_score,
                       const float* __restrict__ w_att,
                       _Float16* __restrict__ wsc16, _Float16* __restrict__ watt16,
                       const float* g1, const float* b1, const float* m1, const float* v1,
                       const float* g2, const float* b2, const float* m2, const float* v2,
                       const float* ga, const float* ba, const float* ma, const float* va,
                       const float* gs, const float* bs, const float* ms, const float* vs,
                       float* __restrict__ bn1, float* __restrict__ bn2,
                       float* __restrict__ bna, float* __restrict__ bns) {
  int i = blockIdx.x * 256 + threadIdx.x;
  if (i < BB_ * KK_ * 128) { colmax[i] = f2k(-3.4e38f); colsum[i] = 0.0f; }
  wsc16[i] = (_Float16)w_score[i];
  watt16[i] = (_Float16)w_att[i];
  if (i < 64) {
    float s = g1[i] * rsqrtf(v1[i] + EPS_);
    bn1[i] = s; bn1[64 + i] = b1[i] - m1[i] * s;
    s = g2[i] * rsqrtf(v2[i] + EPS_);
    bn2[i] = s; bn2[64 + i] = b2[i] - m2[i] * s;
  }
  if (i < 128) {
    float s = ga[i] * rsqrtf(va[i] + EPS_);
    bna[i] = s; bna[128 + i] = ba[i] - ma[i] * s;
    s = gs[i] * rsqrtf(vs[i] + EPS_);
    bns[i] = s; bns[128 + i] = bs[i] - ms[i] * s;
  }
}

// ---------------------------------------------------------------------------
// KNN: one thread per query point, candidates streamed through LDS with
// prefetch of the next tile; register-resident sorted top-16.
// grid = B*N/256 = 64 blocks.
__global__ void k_knn(const float* __restrict__ xyz, int* __restrict__ knn) {
  __shared__ float s_x[512 * 3];
  int tid = threadIdx.x;
  int bb = blockIdx.x >> 5;                // 32 blocks per batch
  int n = (blockIdx.x & 31) * 256 + tid;
  const float* base = xyz + (size_t)bb * NN_ * 3;
  float qx = base[n * 3 + 0], qy = base[n * 3 + 1], qz = base[n * 3 + 2];
  float bd[16]; int bidx[16];
#pragma unroll
  for (int r = 0; r < 16; ++r) { bd[r] = 3.4e38f; bidx[r] = 0; }
  for (int t = 0; t < NN_ / 512; ++t) {
    __syncthreads();
    for (int i = tid; i < 1536; i += 256) s_x[i] = base[t * 1536 + i];
    __syncthreads();
    // prefetch next candidate tile (128B per lane over first 48 lanes)
    if (t + 1 < NN_ / 512 && tid < 48)
      __builtin_prefetch(base + (t + 1) * 1536 + tid * 32, 0, 1);
    for (int j = 0; j < 512; ++j) {
      float dx = s_x[j * 3 + 0] - qx;
      float dy = s_x[j * 3 + 1] - qy;
      float dz = s_x[j * 3 + 2] - qz;
      float d2 = dx * dx + dy * dy + dz * dz;
      if (d2 < bd[15]) {                   // insertion into sorted list
        float cd = d2; int ci = t * 512 + j;
#pragma unroll
        for (int r = 0; r < 16; ++r) {
          if (cd < bd[r]) {
            float td = bd[r]; bd[r] = cd; cd = td;
            int ti = bidx[r]; bidx[r] = ci; ci = ti;
          }
        }
      }
    }
  }
  int* o = knn + ((size_t)bb * NN_ + n) * KK_;
#pragma unroll
  for (int r = 0; r < 16; ++r) o[r] = bidx[r];
}

// ---------------------------------------------------------------------------
// encode: one wave per point.  Builds spatial(16x10), runs two WMMA matmuls
// with folded BN+ReLU -> encoded, gathers neighbor feats, writes concat f16;
// also computes the shortcut branch.  grid = B*N/8 = 2048 blocks x 256.
__global__ void k_encode(const float* __restrict__ xyz,
                         const float* __restrict__ feat,
                         const float* __restrict__ w_loc1,
                         const float* __restrict__ bn1,
                         const float* __restrict__ w_loc2,
                         const float* __restrict__ bn2,
                         const float* __restrict__ w_sc,
                         const float* __restrict__ bns,
                         const int* __restrict__ knn,
                         _Float16* __restrict__ concat,
                         float* __restrict__ shortcut) {
  __shared__ __align__(16) _Float16 s_w1[64 * 32];   // w_loc1 padded 10->32
  __shared__ __align__(16) _Float16 s_w2[64 * 64];
  __shared__ __align__(16) _Float16 s_spat[8][16 * 32];
  __shared__ __align__(16) _Float16 s_h[8][16 * 64];

  int tid = threadIdx.x;
  for (int i = tid; i < 64 * 32; i += 256) {
    int o = i >> 5, c = i & 31;
    s_w1[i] = (c < 10) ? (_Float16)w_loc1[o * 10 + c] : (_Float16)0.0f;
  }
  for (int i = tid; i < 64 * 64; i += 256) s_w2[i] = (_Float16)w_loc2[i];
  __syncthreads();

  int wave = tid >> 5, lane = tid & 31, hl = lane >> 4;
  int gp = blockIdx.x * 8 + wave;          // global point 0..B*N-1
  int bb = gp >> 13;                       // N = 8192

  const float* q = xyz + (size_t)gp * 3;
  float qx = q[0], qy = q[1], qz = q[2];

  int j = lane & 15;
  int nidx = knn[(size_t)gp * KK_ + j];
  const float* pnb = xyz + ((size_t)bb * NN_ + nidx) * 3;
  float nx = pnb[0], ny = pnb[1], nz = pnb[2];
  float rx = nx - qx, ry = ny - qy, rz = nz - qz;
  float dd = rx * rx + ry * ry + rz * rz;

  if (hl == 0) {
    float sp[10] = {qx, qy, qz, nx, ny, nz, rx, ry, rz, dd};
    _Float16* row = &s_spat[wave][j * 32];
#pragma unroll
    for (int c = 0; c < 10; ++c) row[c] = (_Float16)sp[c];
#pragma unroll
    for (int c = 10; c < 32; ++c) row[c] = (_Float16)0.0f;
  }

  // gather neighbor features -> concat cols 64..127 (lane pair splits 64 ch)
  {
    const float* pf = feat + ((size_t)bb * NN_ + nidx) * 64 + hl * 32;
    _Float16* dst = concat + ((size_t)gp * KK_ + j) * 128 + 64 + hl * 32;
#pragma unroll
    for (int c = 0; c < 32; ++c) dst[c] = (_Float16)pf[c];
  }

  // matmul1: h = relu(bn1(spatial @ w1^T))  (16x32pad x 32x64)
  v16h a1 = load_frag_a(&s_spat[wave][0], 32);
#pragma unroll
  for (int nt = 0; nt < 4; ++nt) {
    v16h bf = load_frag_b(&s_w1[nt * 16 * 32], 32);
    v8f d = {};
    d = wmma_f16(a1, bf, d);
    int o = nt * 16 + (lane & 15);
    float sc = bn1[o], bi = bn1[64 + o];
#pragma unroll
    for (int r = 0; r < 8; ++r) {
      int M = r + 8 * hl;
      s_h[wave][M * 64 + o] = (_Float16)fmaxf(d[r] * sc + bi, 0.0f);
    }
  }

  // matmul2: encoded = relu(bn2(h @ w2^T)) -> concat cols 0..63
  v16h a20 = load_frag_a(&s_h[wave][0], 64);
  v16h a21 = load_frag_a(&s_h[wave][32], 64);
#pragma unroll
  for (int nt = 0; nt < 4; ++nt) {
    v8f d = {};
    d = wmma_f16(a20, load_frag_b(&s_w2[nt * 16 * 64], 64), d);
    d = wmma_f16(a21, load_frag_b(&s_w2[nt * 16 * 64 + 32], 64), d);
    int o = nt * 16 + (lane & 15);
    float sc = bn2[o], bi = bn2[64 + o];
#pragma unroll
    for (int r = 0; r < 8; ++r) {
      int M = r + 8 * hl;
      concat[((size_t)gp * KK_ + M) * 128 + o] =
          (_Float16)fmaxf(d[r] * sc + bi, 0.0f);
    }
  }

  // shortcut branch: 128x64 matvec + folded BN (each lane owns 4 channels)
  {
    int o0 = lane * 4;
    float acc[4] = {0.f, 0.f, 0.f, 0.f};
    const float* fr = feat + (size_t)gp * 64;
    for (int c = 0; c < 64; ++c) {
      float f = fr[c];
#pragma unroll
      for (int t = 0; t < 4; ++t) acc[t] += f * w_sc[(o0 + t) * 64 + c];
    }
    float4 r4;
    float* rr = (float*)&r4;
#pragma unroll
    for (int t = 0; t < 4; ++t) {
      int o = o0 + t;
      rr[t] = acc[t] * bns[o] + bns[128 + o];
    }
    ((float4*)(shortcut + (size_t)gp * 128))[lane] = r4;
  }
}

// ---------------------------------------------------------------------------
// softmax-over-N statistics.  mode 0: column max (keyed atomics).
// mode 1: column sum of exp(logit - max).  Logits recomputed via WMMA with
// w_score staged in LDS.  grid = B*(N/64) = 256 blocks x 256.
__global__ void k_stats(const _Float16* __restrict__ concat,
                        const _Float16* __restrict__ wsc16,
                        unsigned* __restrict__ colmax,
                        float* __restrict__ colsum, int mode) {
  __shared__ __align__(16) _Float16 s_wsc[128 * 128];  // 32KB
  __shared__ __align__(16) unsigned s_acc[2048];       // [k=16][c=128]
  __shared__ __align__(16) float s_cmax[2048];
  int tid = threadIdx.x, lane = tid & 31, wave = tid >> 5, hl = lane >> 4;
  int bb = blockIdx.x >> 7;                  // 128 blocks per batch
  int nbase = (blockIdx.x & 127) * 64;

  for (int i = tid; i < 2048; i += 256)
    ((uint4*)s_wsc)[i] = ((const uint4*)wsc16)[i];
  for (int i = tid; i < 2048; i += 256) {
    if (mode == 0) {
      s_acc[i] = f2k(-3.4e38f);
    } else {
      ((float*)s_acc)[i] = 0.0f;
      s_cmax[i] = k2f(colmax[bb * 2048 + i]);
    }
  }
  __syncthreads();

  for (int pi = 0; pi < 8; ++pi) {
    int pt = nbase + wave * 8 + pi;
    const _Float16* cb = concat + ((size_t)bb * NN_ + pt) * (KK_ * 128);
    v16h a0 = load_frag_a(cb, 128);
    v16h a1 = load_frag_a(cb + 32, 128);
    v16h a2 = load_frag_a(cb + 64, 128);
    v16h a3 = load_frag_a(cb + 96, 128);
#pragma unroll
    for (int nt = 0; nt < 8; ++nt) {
      v8f d = {};
      d = wmma_f16(a0, load_frag_b(&s_wsc[nt * 2048], 128), d);
      d = wmma_f16(a1, load_frag_b(&s_wsc[nt * 2048 + 32], 128), d);
      d = wmma_f16(a2, load_frag_b(&s_wsc[nt * 2048 + 64], 128), d);
      d = wmma_f16(a3, load_frag_b(&s_wsc[nt * 2048 + 96], 128), d);
      int c = nt * 16 + (lane & 15);
#pragma unroll
      for (int r = 0; r < 8; ++r) {
        int k = r + 8 * hl;
        if (mode == 0)
          atomicMax(&s_acc[k * 128 + c], f2k(d[r]));
        else
          atomicAdd((float*)&s_acc[k * 128 + c], __expf(d[r] - s_cmax[k * 128 + c]));
      }
    }
  }
  __syncthreads();
  for (int i = tid; i < 2048; i += 256) {
    if (mode == 0)
      atomicMax(&colmax[bb * 2048 + i], s_acc[i]);
    else
      atomicAdd(&colsum[bb * 2048 + i], ((float*)s_acc)[i]);
  }
}

// ---------------------------------------------------------------------------
// final: per point, TDM-stage concat tile to LDS, recompute logits (WMMA),
// scores, pooled; then per-wave 16-point att matmul (WMMA) + folded BN +
// shortcut + ReLU.  grid = B*(N/128) = 128 blocks x 256 (8 waves x 16 pts).
__global__ void k_final(const _Float16* __restrict__ concat,
                        const _Float16* __restrict__ wsc16,
                        const _Float16* __restrict__ watt16,
                        const unsigned* __restrict__ colmax,
                        const float* __restrict__ colsum,
                        const float* __restrict__ bna,
                        const float* __restrict__ shortcut,
                        float* __restrict__ out) {
  __shared__ __align__(16) _Float16 s_ct[8][16 * 128];    // one point's concat
  __shared__ __align__(16) _Float16 s_pool[8][16 * 128];  // 16 points' pooled
  int tid = threadIdx.x, lane = tid & 31, wave = tid >> 5, hl = lane >> 4;
  int bb = blockIdx.x >> 6;                 // 64 blocks per batch
  int nbase = (blockIdx.x & 63) * 128 + wave * 16;
  const unsigned* cmaxb = colmax + bb * 2048;
  const float* csumb = colsum + bb * 2048;

  for (int p = 0; p < 16; ++p) {
    size_t gp = (size_t)bb * NN_ + nbase + p;
    // async DMA this point's 16x128 f16 concat tile into LDS (per-wave TDM)
    tdm_load_tile_f16(&s_ct[wave][0], concat + gp * (KK_ * 128), KK_ * 128);

    v16h a0 = load_frag_a(&s_ct[wave][0], 128);
    v16h a1 = load_frag_a(&s_ct[wave][32], 128);
    v16h a2 = load_frag_a(&s_ct[wave][64], 128);
    v16h a3 = load_frag_a(&s_ct[wave][96], 128);
#pragma unroll
    for (int nt = 0; nt < 8; ++nt) {
      v8f d = {};
      d = wmma_f16(a0, load_frag_b(wsc16 + nt * 2048, 128), d);
      d = wmma_f16(a1, load_frag_b(wsc16 + nt * 2048 + 32, 128), d);
      d = wmma_f16(a2, load_frag_b(wsc16 + nt * 2048 + 64, 128), d);
      d = wmma_f16(a3, load_frag_b(wsc16 + nt * 2048 + 96, 128), d);
      int c = nt * 16 + (lane & 15);
      float psum = 0.0f;
#pragma unroll
      for (int r = 0; r < 8; ++r) {
        int k = r + 8 * hl;
        float cm = k2f(cmaxb[k * 128 + c]);
        float cs = csumb[k * 128 + c];
        float scv = __expf(d[r] - cm) / cs;
        psum += (float)s_ct[wave][k * 128 + c] * scv;
      }
      psum += __shfl_xor(psum, 16, 32);     // combine the two M-halves
      if (lane < 16) s_pool[wave][p * 128 + nt * 16 + lane] = (_Float16)psum;
    }
  }

  // att = relu(bn(pooled @ w_att^T)); out = relu(att + shortcut)
  v16h pa0 = load_frag_a(&s_pool[wave][0], 128);
  v16h pa1 = load_frag_a(&s_pool[wave][32], 128);
  v16h pa2 = load_frag_a(&s_pool[wave][64], 128);
  v16h pa3 = load_frag_a(&s_pool[wave][96], 128);
#pragma unroll
  for (int nt = 0; nt < 8; ++nt) {
    v8f d = {};
    d = wmma_f16(pa0, load_frag_b(watt16 + nt * 2048, 128), d);
    d = wmma_f16(pa1, load_frag_b(watt16 + nt * 2048 + 32, 128), d);
    d = wmma_f16(pa2, load_frag_b(watt16 + nt * 2048 + 64, 128), d);
    d = wmma_f16(pa3, load_frag_b(watt16 + nt * 2048 + 96, 128), d);
    int o = nt * 16 + (lane & 15);
    float sc = bna[o], bi = bna[128 + o];
#pragma unroll
    for (int r = 0; r < 8; ++r) {
      int M = r + 8 * hl;
      size_t row = (size_t)bb * NN_ + nbase + M;
      float att = fmaxf(d[r] * sc + bi, 0.0f);
      out[row * 128 + o] = fmaxf(att + shortcut[row * 128 + o], 0.0f);
    }
  }
}

// ---------------------------------------------------------------------------
extern "C" void kernel_launch(void* const* d_in, const int* in_sizes, int n_in,
                              void* d_out, int out_size, void* d_ws, size_t ws_size,
                              hipStream_t stream) {
  (void)in_sizes; (void)n_in; (void)out_size; (void)ws_size;
  const float* xyz    = (const float*)d_in[0];
  const float* feat   = (const float*)d_in[1];
  const float* w_loc1 = (const float*)d_in[2];
  const float* g1 = (const float*)d_in[3];
  const float* b1 = (const float*)d_in[4];
  const float* m1 = (const float*)d_in[5];
  const float* v1 = (const float*)d_in[6];
  const float* w_loc2 = (const float*)d_in[7];
  const float* g2 = (const float*)d_in[8];
  const float* b2 = (const float*)d_in[9];
  const float* m2 = (const float*)d_in[10];
  const float* v2 = (const float*)d_in[11];
  const float* w_score = (const float*)d_in[12];
  const float* w_att   = (const float*)d_in[13];
  const float* ga = (const float*)d_in[14];
  const float* ba = (const float*)d_in[15];
  const float* ma = (const float*)d_in[16];
  const float* va = (const float*)d_in[17];
  const float* w_sc = (const float*)d_in[18];
  const float* gs = (const float*)d_in[19];
  const float* bs = (const float*)d_in[20];
  const float* ms = (const float*)d_in[21];
  const float* vs = (const float*)d_in[22];
  float* out = (float*)d_out;

  char* ws = (char*)d_ws;
  size_t off = 0;
  int* ws_knn = (int*)(ws + off);              off += (size_t)BB_ * NN_ * KK_ * 4;
  _Float16* ws_concat = (_Float16*)(ws + off); off += (size_t)BB_ * NN_ * KK_ * 128 * 2;
  float* ws_short = (float*)(ws + off);        off += (size_t)BB_ * NN_ * 128 * 4;
  unsigned* ws_cmax = (unsigned*)(ws + off);   off += (size_t)BB_ * KK_ * 128 * 4;
  float* ws_csum = (float*)(ws + off);         off += (size_t)BB_ * KK_ * 128 * 4;
  _Float16* wsc16 = (_Float16*)(ws + off);     off += 128 * 128 * 2;
  _Float16* watt16 = (_Float16*)(ws + off);    off += 128 * 128 * 2;
  float* ws_bn1 = (float*)(ws + off);          off += 128 * 4;
  float* ws_bn2 = (float*)(ws + off);          off += 128 * 4;
  float* ws_bna = (float*)(ws + off);          off += 256 * 4;
  float* ws_bns = (float*)(ws + off);          off += 256 * 4;

  k_init<<<64, 256, 0, stream>>>(ws_cmax, ws_csum, w_score, w_att, wsc16, watt16,
                                 g1, b1, m1, v1, g2, b2, m2, v2,
                                 ga, ba, ma, va, gs, bs, ms, vs,
                                 ws_bn1, ws_bn2, ws_bna, ws_bns);
  k_knn<<<BB_ * NN_ / 256, 256, 0, stream>>>(xyz, ws_knn);
  k_encode<<<BB_ * NN_ / 8, 256, 0, stream>>>(
      xyz, feat, w_loc1, ws_bn1, w_loc2, ws_bn2, w_sc, ws_bns,
      ws_knn, ws_concat, ws_short);
  k_stats<<<BB_ * (NN_ / 64), 256, 0, stream>>>(ws_concat, wsc16, ws_cmax, ws_csum, 0);
  k_stats<<<BB_ * (NN_ / 64), 256, 0, stream>>>(ws_concat, wsc16, ws_cmax, ws_csum, 1);
  k_final<<<BB_ * (NN_ / 128), 256, 0, stream>>>(
      ws_concat, wsc16, watt16, ws_cmax, ws_csum, ws_bna, ws_short, out);
}
